// DecoderLayer_25847113187504
// MI455X (gfx1250) — matrix operand order repaired and checked by reading, not compiled, not run on hardware
//
#include <hip/hip_runtime.h>
#include <cstdint>
#include <cstddef>

// ---------------------------------------------------------------------------
// CDNA5 (gfx1250) WMMA types
// ---------------------------------------------------------------------------
typedef __attribute__((ext_vector_type(16))) __bf16 v16bf;
typedef __attribute__((ext_vector_type(2)))  __bf16 v2bf;
typedef __attribute__((ext_vector_type(2)))  float  v2f;
typedef __attribute__((ext_vector_type(8)))  float  v8f;

union FragBF {
  v16bf v;
  unsigned w[8];
  unsigned short u[16];
};

__device__ __forceinline__ unsigned pack2bf(float lo, float hi) {
  // f32x2 -> bf16x2; lowers to v_cvt_pk_bf16_f32 when the target has it.
  v2f f = {lo, hi};
  v2bf r = __builtin_convertvector(f, v2bf);
  return __builtin_bit_cast(unsigned, r);
}

__device__ __forceinline__ float gelu_tanh(float v) {
  float u = 0.7978845608028654f * (v + 0.044715f * v * v * v);
  return 0.5f * v * (1.0f + tanhf(u));
}

// ---------------------------------------------------------------------------
// GEMM: C = A(MxK) @ B(KxN), bf16 WMMA, f32 accumulate.
// One wave computes a 32x64 tile = 2(M) x 4(N) WMMA tiles; K stepped by 32.
// LDB is a compile-time constant so B gathers use immediate offsets.
// modes: 0 none; 1 +bias; 2 gelu(x+bias); 3 +bias+R; 4 +R
// Output row remap: orow = (m % segIn) + (m / segIn) * segFull + segOff
//   (identity: segIn = M, segOff = 0)
// ---------------------------------------------------------------------------
template <int LDB, bool NGUARD>
__global__ void __launch_bounds__(32)
gemm_bf16_wmma(const float* __restrict__ A, const float* __restrict__ B,
               const float* __restrict__ bias, const float* __restrict__ R,
               float* __restrict__ C,
               int N, int K, int lda, int ldc, int ldr,
               int segIn, int segFull, int segOff, int mode) {
  const int lane = threadIdx.x;
  const int half = lane >> 4;   // 0: lanes 0-15, 1: lanes 16-31
  const int lm   = lane & 15;
  const int m0   = blockIdx.y << 5;   // 2 M-tiles of 16
  const int n0   = blockIdx.x << 6;   // 4 N-tiles of 16

  v8f acc[2][4] = {};
  // A fragment (16-bit A 16x32 layout): lane half selects K sub-block.
  // j in [0,8): k = k0 + half*8 + j ; j in [8,16): k = k0 + 16 + half*8 + (j-8)
  const float* Arow0 = A + (size_t)(m0 + lm) * lda + (half << 3);
  const float* Arow1 = Arow0 + (size_t)16 * lda;

  for (int k0 = 0; k0 < K; k0 += 32) {
    FragBF fa0, fa1;
#pragma unroll
    for (int p = 0; p < 4; ++p) {
      fa0.w[p]     = pack2bf(Arow0[k0 + 2*p],      Arow0[k0 + 2*p + 1]);
      fa0.w[4 + p] = pack2bf(Arow0[k0 + 16 + 2*p], Arow0[k0 + 16 + 2*p + 1]);
      fa1.w[p]     = pack2bf(Arow1[k0 + 2*p],      Arow1[k0 + 2*p + 1]);
      fa1.w[4 + p] = pack2bf(Arow1[k0 + 16 + 2*p], Arow1[k0 + 16 + 2*p + 1]);
    }
    // B fragment (32x16): j in [0,8): k = k0 + half*8 + j ; j in [8,16): k = k0 + 16 + half*8 + (j-8), column n.
    const float* bbase = B + (size_t)(k0 + (half << 3)) * LDB + n0 + lm;
#pragma unroll
    for (int tb = 0; tb < 4; ++tb) {
      FragBF fb;
      if (!NGUARD || (n0 + 16*tb + lm) < N) {
        const float* bp = bbase + 16*tb;
#pragma unroll
        for (int p = 0; p < 8; ++p)
          fb.w[p] = pack2bf(bp[(2*p + (p >> 2) * 8) * LDB], bp[(2*p + (p >> 2) * 8 + 1) * LDB]);
      } else {
#pragma unroll
        for (int p = 0; p < 8; ++p) fb.w[p] = 0u;
      }
      // EXEC is all ones here (guards only shape register contents).
      acc[0][tb] = __builtin_amdgcn_wmma_f32_16x16x32_bf16(
          false, fa0.v, false, fb.v, (short)0, acc[0][tb], false, false);
      acc[1][tb] = __builtin_amdgcn_wmma_f32_16x16x32_bf16(
          false, fa1.v, false, fb.v, (short)0, acc[1][tb], false, false);
    }
  }

#pragma unroll
  for (int tm = 0; tm < 2; ++tm) {
#pragma unroll
    for (int tb = 0; tb < 4; ++tb) {
      int nt = n0 + 16*tb + lm;
      if (NGUARD && nt >= N) continue;
      float bv = (mode == 1 || mode == 2 || mode == 3) ? bias[nt] : 0.0f;
#pragma unroll
      for (int g = 0; g < 8; ++g) {
        int m = m0 + (tm << 4) + (half << 3) + g;   // C layout: row = 8*half+g
        int orow = (m % segIn) + (m / segIn) * segFull + segOff;
        float v = acc[tm][tb][g] + bv;
        if (mode == 2) v = gelu_tanh(v);
        if (mode == 3 || mode == 4) v += R[(size_t)orow * ldr + nt];
        C[(size_t)orow * ldc + nt] = v;
      }
    }
  }
}

// ---------------------------------------------------------------------------
// Depthwise causal conv (width 4) + SiLU over (Nseq, L, 512) row-major.
// ---------------------------------------------------------------------------
__global__ void dwconv_silu(const float* __restrict__ xc,
                            const float* __restrict__ w,
                            const float* __restrict__ b,
                            float* __restrict__ xs, int Nseq, int L) {
  int idx = blockIdx.x * blockDim.x + threadIdx.x;
  int total = Nseq * L * 512;
  if (idx >= total) return;
  int d  = idx & 511;
  int nt = idx >> 9;            // n*L + t
  int t  = nt % L;
  float acc = b[d];
  const float* base = xc + (size_t)(nt - t) * 512 + d;   // sequence start
#pragma unroll
  for (int i = 0; i < 4; ++i) {
    int tt = t - 3 + i;
    if (tt >= 0) acc += w[d * 4 + i] * base[(size_t)tt * 512];
  }
  xs[(size_t)idx] = acc / (1.0f + __expf(-acc));
}

// ---------------------------------------------------------------------------
// Fused selective-scan: dt projection (K=16) + softplus, SSM recurrence over
// L, + x*D skip, * SiLU(z) gate.  One thread per (sequence, channel).
// Output rows are compacted to t >= tSkip (cross-attention case).
// ---------------------------------------------------------------------------
__global__ void ssm_scan(const float* __restrict__ dbl,   // (Nseq*L, 48)
                         const float* __restrict__ xs,    // (Nseq*L, 512)
                         const float* __restrict__ zb,    // (Nseq*(L-tSkip), 512)
                         const float* __restrict__ Wdt,   // (16, 512)
                         const float* __restrict__ bdt,   // (512)
                         const float* __restrict__ A_log, // (512, 16)
                         const float* __restrict__ Dp,    // (512)
                         float* __restrict__ y,           // (Nseq*(L-tSkip), 512)
                         int Nseq, int L, int tSkip) {
  int idx = blockIdx.x * blockDim.x + threadIdx.x;
  if (idx >= Nseq * 512) return;
  int d = idx & 511;
  int n = idx >> 9;
  float wdt[16], a[16], h[16];
#pragma unroll
  for (int s = 0; s < 16; ++s) {
    wdt[s] = Wdt[s * 512 + d];
    a[s]   = -__expf(A_log[d * 16 + s]);
    h[s]   = 0.0f;
  }
  const float bd = bdt[d], Dd = Dp[d];
  const int Lout = L - tSkip;
  for (int t = 0; t < L; ++t) {
    const float* row = dbl + (size_t)(n * L + t) * 48;
    float dtr = bd;
#pragma unroll
    for (int r = 0; r < 16; ++r) dtr += row[r] * wdt[r];
    float dt = (dtr > 20.0f) ? dtr : log1pf(__expf(dtr));  // softplus
    float x  = xs[(size_t)(n * L + t) * 512 + d];
    float dx = dt * x;
    float yacc = 0.0f;
#pragma unroll
    for (int s = 0; s < 16; ++s) {
      h[s] = h[s] * __expf(dt * a[s]) + dx * row[16 + s];  // B_t
      yacc += h[s] * row[32 + s];                          // C_t
    }
    if (t >= tSkip) {
      size_t o = ((size_t)n * Lout + (t - tSkip)) * 512 + d;
      float z  = zb[o];
      float yv = yacc + x * Dd;
      y[o] = yv * (z / (1.0f + __expf(-z)));
    }
  }
}

// ---------------------------------------------------------------------------
// RMSNorm over rows of 256, optional transposed store:
//   row = bb*L1*L2 + i*L2 + j  ->  orow = bb*L1*L2 + j*L1 + i
// ---------------------------------------------------------------------------
__global__ void rmsnorm256(const float* __restrict__ in,
                           const float* __restrict__ w,
                           float* __restrict__ out,
                           int L1, int L2, int permute) {
  int row = blockIdx.x;
  int tid = threadIdx.x;
  float v = in[(size_t)row * 256 + tid];
  float p = v * v;
#pragma unroll
  for (int off = 16; off > 0; off >>= 1) p += __shfl_xor(p, off, 32);
  __shared__ float red[8];
  __shared__ float stot;
  if ((tid & 31) == 0) red[tid >> 5] = p;
  __syncthreads();
  if (tid == 0) {
    float s = 0.0f;
#pragma unroll
    for (int i = 0; i < 8; ++i) s += red[i];
    stot = s;
  }
  __syncthreads();
  float scale = rsqrtf(stot * (1.0f / 256.0f) + 1e-5f);
  int orow = row;
  if (permute) {
    int bb  = row / (L1 * L2);
    int rem = row - bb * L1 * L2;
    int i = rem / L2, j = rem - i * L2;
    orow = bb * L1 * L2 + j * L1 + i;
  }
  out[(size_t)orow * 256 + tid] = v * scale * w[tid];
}

// ---------------------------------------------------------------------------
// Host side
// ---------------------------------------------------------------------------
struct MambaP { const float *A_log, *D, *Wdt, *Win, *Wout, *Wx, *bdt, *convb, *convw; };
struct MlpP   { const float *W1, *W2, *b1, *b2; };

// Workspace layout (float offsets).  Regions reused across stages:
// OFF_XC hosts xc, then y (scan output), then MLP hidden.
static constexpr size_t OFF_XC    = 0;           // 73728*512  = 37,748,736
static constexpr size_t OFF_ZB    = 37748736;    // 24576*512  = 12,582,912
static constexpr size_t OFF_XS    = 50331648;    // 73728*512  = 37,748,736
static constexpr size_t OFF_DBL   = 88080384;    // 73728*48   =  3,538,944
static constexpr size_t OFF_T1    = 91619328;    // 24576*256  =  6,291,456
static constexpr size_t OFF_Z1    = 97910784;    // 24576*256
static constexpr size_t OFF_DIMIN = 104202240;   // 24576*256
static constexpr size_t OFF_ZDEC  = 110493696;   // 24576*256
static constexpr size_t OFF_ZS    = 116785152;   // 24576*256
static constexpr size_t OFF_ZTMP  = 123076608;   // 24576*256
static constexpr size_t WS_FLOATS = 129368064;   // ~494 MB

// Only the (LDB, NGUARD) combos that are actually launched are instantiated:
// (1024,false), (256,false), (48,true), (24,true).
static inline void launch_gemm(hipStream_t s, const float* A, const float* B,
                               const float* bias, const float* R, float* C,
                               int M, int N, int K, int lda, int ldb, int ldc, int ldr,
                               int segIn, int segFull, int segOff, int mode) {
  dim3 grid((N + 63) / 64, M / 32);
  if (ldb == 1024) {
    gemm_bf16_wmma<1024, false><<<grid, 32, 0, s>>>(A, B, bias, R, C, N, K,
        lda, ldc, ldr, segIn, segFull, segOff, mode);
  } else if (ldb == 256) {
    gemm_bf16_wmma<256, false><<<grid, 32, 0, s>>>(A, B, bias, R, C, N, K,
        lda, ldc, ldr, segIn, segFull, segOff, mode);
  } else if (ldb == 48) {
    gemm_bf16_wmma<48, true><<<grid, 32, 0, s>>>(A, B, bias, R, C, N, K,
        lda, ldc, ldr, segIn, segFull, segOff, mode);
  } else if (ldb == 24) {
    gemm_bf16_wmma<24, true><<<grid, 32, 0, s>>>(A, B, bias, R, C, N, K,
        lda, ldc, ldr, segIn, segFull, segOff, mode);
  }
}

static void run_mammlp(hipStream_t s, float* ws, const float* X,
                       const MambaP& mp, const MlpP& lp, const float* n1,
                       int Nseq, int L, float* outBuf,
                       int permute, int pL1, int pL2) {
  const int tok = Nseq * L;
  float* XC  = ws + OFF_XC;
  float* ZB  = ws + OFF_ZB;
  float* XS  = ws + OFF_XS;
  float* DBL = ws + OFF_DBL;
  float* T1  = ws + OFF_T1;
  float* Z1  = ws + OFF_Z1;

  // xz = X @ Win, split into xc (cols 0..511) and z (cols 512..1023)
  launch_gemm(s, X, mp.Win,       nullptr, nullptr, XC, tok, 512, 256, 256, 1024, 512, 0, tok, tok, 0, 0);
  launch_gemm(s, X, mp.Win + 512, nullptr, nullptr, ZB, tok, 512, 256, 256, 1024, 512, 0, tok, tok, 0, 0);
  // depthwise causal conv + SiLU
  {
    int total = tok * 512;
    dwconv_silu<<<(total + 255) / 256, 256, 0, s>>>(XC, mp.convw, mp.convb, XS, Nseq, L);
  }
  // dbl = xs @ Wx
  launch_gemm(s, XS, mp.Wx, nullptr, nullptr, DBL, tok, 48, 512, 512, 48, 48, 0, tok, tok, 0, 0);
  // fused scan + gate -> y (reuse XC)
  {
    int total = Nseq * 512;
    ssm_scan<<<(total + 255) / 256, 256, 0, s>>>(DBL, XS, ZB, mp.Wdt, mp.bdt,
                                                 mp.A_log, mp.D, XC, Nseq, L, 0);
  }
  // T1 = y @ Wout + X (residual)
  launch_gemm(s, XC, mp.Wout, nullptr, X, T1, tok, 256, 512, 512, 256, 256, 256, tok, tok, 0, 4);
  rmsnorm256<<<tok, 256, 0, s>>>(T1, n1, Z1, 0, 0, 0);
  // MLP: H = gelu(Z1 @ W1 + b1) (reuse XC); T1 = H @ W2 + b2 + Z1
  launch_gemm(s, Z1, lp.W1, lp.b1, nullptr, XC, tok, 1024, 256, 256, 1024, 1024, 0, tok, tok, 0, 2);
  launch_gemm(s, XC, lp.W2, lp.b2, Z1, T1, tok, 256, 1024, 1024, 256, 256, 256, tok, tok, 0, 3);
  rmsnorm256<<<tok, 256, 0, s>>>(T1, n1, outBuf, pL1, pL2, permute);
}

extern "C" void kernel_launch(void* const* d_in, const int* in_sizes, int n_in,
                              void* d_out, int out_size, void* d_ws, size_t ws_size,
                              hipStream_t stream) {
  (void)in_sizes; (void)out_size;
  if (n_in < 47) return;
  if (ws_size < WS_FLOATS * sizeof(float)) return;

  // setup_inputs() pytree flattened with sorted dict keys:
  const float* E_dec = (const float*)d_in[0];   // (8,64,48,256)
  const float* Z_enc = (const float*)d_in[1];   // (8,64,96,256)
  const float* Wp    = (const float*)d_in[2];
  const float* bp    = (const float*)d_in[3];
  MambaP cr   { (const float*)d_in[4],  (const float*)d_in[5],  (const float*)d_in[6],
                (const float*)d_in[7],  (const float*)d_in[8],  (const float*)d_in[9],
                (const float*)d_in[10], (const float*)d_in[11], (const float*)d_in[12] };
  MlpP   dm   { (const float*)d_in[13], (const float*)d_in[14],
                (const float*)d_in[15], (const float*)d_in[16] };
  MambaP dimm { (const float*)d_in[17], (const float*)d_in[18], (const float*)d_in[19],
                (const float*)d_in[20], (const float*)d_in[21], (const float*)d_in[22],
                (const float*)d_in[23], (const float*)d_in[24], (const float*)d_in[25] };
  MlpP   dlp  { (const float*)d_in[26], (const float*)d_in[27],
                (const float*)d_in[28], (const float*)d_in[29] };
  const float* dim_n1 = (const float*)d_in[30];
  const float* n1_top = (const float*)d_in[31];
  const float* n2_top = (const float*)d_in[32];
  MambaP tm   { (const float*)d_in[33], (const float*)d_in[34], (const float*)d_in[35],
                (const float*)d_in[36], (const float*)d_in[37], (const float*)d_in[38],
                (const float*)d_in[39], (const float*)d_in[40], (const float*)d_in[41] };
  MlpP   tlp  { (const float*)d_in[42], (const float*)d_in[43],
                (const float*)d_in[44], (const float*)d_in[45] };
  const float* time_n1 = (const float*)d_in[46];

  float* ws = (float*)d_ws;
  float* DIMIN = ws + OFF_DIMIN;
  float* ZDEC  = ws + OFF_ZDEC;

  // Stage 1: time mammlp on (512 seqs, L=48); transpose (b,oc,ts)->(b,ts,oc)
  run_mammlp(stream, ws, E_dec, tm, tlp, time_n1, 512, 48, DIMIN, 1, 64, 48);
  // Stage 2: dim mammlp on (384 seqs, L=64); transpose back -> Z_dec
  run_mammlp(stream, ws, DIMIN, dimm, dlp, dim_n1, 384, 64, ZDEC, 1, 48, 64);

  // Stage 3: cross mamba on virtual concat [Z_enc(96) ; Z_dec(48)] per seq.
  {
    const int tokc = 512 * 144;
    float* XC  = ws + OFF_XC;
    float* ZB  = ws + OFF_ZB;
    float* XS  = ws + OFF_XS;
    float* DBL = ws + OFF_DBL;
    float* T1  = ws + OFF_T1;
    float* ZS  = ws + OFF_ZS;
    float* ZTMP= ws + OFF_ZTMP;

    // xc for full concat sequence, via segmented output-row remap (no Z_cat).
    launch_gemm(stream, Z_enc, cr.Win, nullptr, nullptr, XC,
                512 * 96, 512, 256, 256, 1024, 512, 0, 96, 144, 0, 0);
    launch_gemm(stream, ZDEC, cr.Win, nullptr, nullptr, XC,
                512 * 48, 512, 256, 256, 1024, 512, 0, 48, 144, 96, 0);
    // z gate only needed for the 48 decoder tokens (t >= 96), compacted.
    launch_gemm(stream, ZDEC, cr.Win + 512, nullptr, nullptr, ZB,
                512 * 48, 512, 256, 256, 1024, 512, 0, 24576, 24576, 0, 0);
    dwconv_silu<<<(tokc * 512 + 255) / 256, 256, 0, stream>>>(XC, cr.convw, cr.convb, XS, 512, 144);
    launch_gemm(stream, XS, cr.Wx, nullptr, nullptr, DBL,
                tokc, 48, 512, 512, 48, 48, 0, tokc, tokc, 0, 0);
    // scan over full 144 tokens, emit gated y only for t>=96 (compacted)
    ssm_scan<<<(512 * 512 + 255) / 256, 256, 0, stream>>>(DBL, XS, ZB, cr.Wdt, cr.bdt,
                                                          cr.A_log, cr.D, XC, 512, 144, 96);
    // ZS = y @ Wout + Z_dec ; Z_tmp = rmsnorm(ZS, n1)
    launch_gemm(stream, XC, cr.Wout, nullptr, ZDEC, ZS,
                24576, 256, 512, 512, 256, 256, 256, 24576, 24576, 0, 4);
    rmsnorm256<<<24576, 256, 0, stream>>>(ZS, n1_top, ZTMP, 0, 0, 0);
    // dec_mlp (256->256->256) with residual, then rmsnorm(n2) -> dec output
    launch_gemm(stream, ZTMP, dm.W1, dm.b1, nullptr, XC,
                24576, 256, 256, 256, 256, 256, 0, 24576, 24576, 0, 2);
    launch_gemm(stream, XC, dm.W2, dm.b2, ZTMP, T1,
                24576, 256, 256, 256, 256, 256, 256, 24576, 24576, 0, 3);
    float* dec = (float*)d_out;                 // dec_output: 24576 x 256
    rmsnorm256<<<24576, 256, 0, stream>>>(T1, n2_top, dec, 0, 0, 0);
    // layer_predict = dec @ Wp + bp : 24576 x 24 (N-edge masked in kernel)
    launch_gemm(stream, dec, Wp, bp, nullptr, dec + 6291456,
                24576, 24, 256, 256, 24, 24, 0, 24576, 24576, 0, 1);
  }
}
